// TDDNetwork_41489384080001
// MI455X (gfx1250) — matrix operand order repaired and checked by reading, not compiled, not run on hardware
//
#include <hip/hip_runtime.h>
#include <hip/hip_bf16.h>

typedef __attribute__((ext_vector_type(16))) _Float16 v16h;
typedef __attribute__((ext_vector_type(8)))  _Float16 h8;
typedef __attribute__((ext_vector_type(8)))  float    v8f;

// ---------------------------------------------------------------------------
// Packed-f16 GEMM: one wave computes a (16*MT) x (16*NT) output tile with
// v_wmma_f32_16x16x32_f16. A and B are loaded as CONTIGUOUS f16 runs:
//   A (16x32 f16): lane = 16g+mr ; elems 0..7  -> K = k0+16g .. +7   (8-half run)
//                                 elems 8..15 -> K = k0+16g+8 .. +7 (8-half run)
//   B (32x16 f16): lane = 16g+nc ; elems 0..15 -> K = k0+16g .. +15 (16-half run)
//   C/D f32: VGPR v -> M = v + 8g, N = lane&15
// aptr(mrow,k) / bptr(n,k) return pointers to those runs; store(mi,n,acc).
// B fragments are reused across the MT M-subtiles.
// ---------------------------------------------------------------------------
template <int MT, int NT, typename FA, typename FB, typename FS>
__device__ __forceinline__ void wmma_gemm_tile(int K, FA aptr, FB bptr, FS store) {
  const int lane = threadIdx.x & 31;
  const int g    = lane >> 4;
  const int mr   = lane & 15;
  const int nc   = lane & 15;
  v8f acc[MT][NT];
#pragma unroll
  for (int m = 0; m < MT; ++m)
#pragma unroll
    for (int t = 0; t < NT; ++t) acc[m][t] = (v8f){};

  for (int k0 = 0; k0 < K; k0 += 32) {
    v16h a[MT];
#pragma unroll
    for (int m = 0; m < MT; ++m) {
      const h8 r0 = *(const h8*)aptr(m * 16 + mr, k0 + 16 * g);
      const h8 r1 = *(const h8*)aptr(m * 16 + mr, k0 + 16 * g + 8);
#pragma unroll
      for (int e = 0; e < 8; ++e) { a[m][e] = r0[e]; a[m][8 + e] = r1[e]; }
    }
#pragma unroll
    for (int t = 0; t < NT; ++t) {
      const v16h b = *(const v16h*)bptr(t * 16 + nc, k0 + 16 * g);
#pragma unroll
      for (int m = 0; m < MT; ++m)
        acc[m][t] = __builtin_amdgcn_wmma_f32_16x16x32_f16(
            false, a[m], false, b, (short)0, acc[m][t], false, false);
    }
  }
#pragma unroll
  for (int m = 0; m < MT; ++m)
#pragma unroll
    for (int t = 0; t < NT; ++t)
#pragma unroll
      for (int v = 0; v < 8; ++v)
        store(m * 16 + v + 8 * g, t * 16 + nc, acc[m][t][v]);
}

// ---------------------------------------------------------------------------
// Input packing: NCHW fp32 [2][512][4][84][84] -> NHWC f16 [2][512][84][84][4]
// ---------------------------------------------------------------------------
__global__ void pack_in_kernel(const float* __restrict__ curr,
                               const float* __restrict__ next,
                               _Float16* __restrict__ o) {
  const size_t i = (size_t)blockIdx.x * 256 + threadIdx.x;  // 28,901,376 total
  if (i < 28901376u) {
    size_t rem = i;
    const int c = (int)(rem & 3); rem >>= 2;
    const int x = (int)(rem % 84); rem /= 84;
    const int y = (int)(rem % 84); rem /= 84;
    const int b = (int)(rem % 512);
    const int z = (int)(rem / 512);
    const float* in = z ? next : curr;
    o[i] = (_Float16)in[((size_t)(b * 4 + c) * 84 + y) * 84 + x];
  }
}

// ---------------------------------------------------------------------------
// Weight packing kernels (fp32 -> f16, [N][K] row-major, K in (pos, c) order)
// ---------------------------------------------------------------------------
// conv weights OIHW [O][C][P] -> [O][P*C] with k' = pos*C + c
__global__ void pack_convw_kernel(const float* __restrict__ w, _Float16* __restrict__ o,
                                  int O, int C, int P) {
  const int i = blockIdx.x * 256 + threadIdx.x;
  const int K = C * P;
  if (i < O * K) {
    const int oc = i / K, k = i % K;
    const int pos = k / C, c = k % C;
    o[i] = (_Float16)w[(size_t)oc * K + c * P + pos];
  }
}

// fcw [3136(NCHW k)][256] -> [256][3136] with k' = pos*64 + c (NHWC order)
__global__ void pack_fcw_kernel(const float* __restrict__ w, _Float16* __restrict__ o) {
  const int i = blockIdx.x * 256 + threadIdx.x;
  if (i < 256 * 3136) {
    const int n = i / 3136, k = i % 3136;
    const int pos = k >> 6, c = k & 63;
    o[i] = (_Float16)w[(size_t)(c * 49 + pos) * 256 + n];
  }
}

// dense [IN][OUT] -> [OUT][IN]
__global__ void packT_kernel(const float* __restrict__ w, _Float16* __restrict__ o,
                             int IN, int OUT) {
  const int i = blockIdx.x * 256 + threadIdx.x;
  if (i < IN * OUT) {
    const int n = i / IN, k = i % IN;
    o[i] = (_Float16)w[(size_t)k * OUT + n];
  }
}

// ---------------------------------------------------------------------------
// conv1: NHWC f16 [512,84,84,4], 8x8 s4 -> NHWC f16 [512,400,32]
// M=204800 (32-row tiles), N=32 (NT=2), K=256, k' = (ky*8+kx)*4 + c
// 8-half A-runs cover 2 adjacent pixels (kx even) -> contiguous, 16B aligned.
// ---------------------------------------------------------------------------
__global__ __launch_bounds__(32)
void conv1_kernel(const _Float16* __restrict__ inp_all, const _Float16* __restrict__ wp,
                  const float* __restrict__ bias, _Float16* __restrict__ out) {
  const _Float16* in = inp_all + (size_t)blockIdx.z * (512u * 7056u * 4u);
  _Float16* o = out + (size_t)blockIdx.z * (512u * 400u * 32u);
  const int m0 = blockIdx.x * 32;
  wmma_gemm_tile<2, 2>(256,
      [&](int mrow, int k) -> const _Float16* {
        const int m = m0 + mrow;
        const int b = m / 400, rem = m % 400;
        const int oy = rem / 20, ox = rem % 20;
        const int pos = k >> 2, c = k & 3;
        const int ky = pos >> 3, kx = pos & 7;
        return in + (((size_t)b * 84 + (oy * 4 + ky)) * 84 + (ox * 4 + kx)) * 4 + c;
      },
      [&](int n, int k) -> const _Float16* { return wp + (size_t)n * 256 + k; },
      [&](int mi, int n, float av) {
        const int m = m0 + mi;
        float val = av + bias[n];
        val = val > 0.f ? val : 0.f;
        o[(size_t)m * 32 + n] = (_Float16)val;
      });
}

// ---------------------------------------------------------------------------
// conv2: NHWC f16 [512,20,20,32], 4x4 s2 -> NHWC f16 [512,81,64]
// M=41472 (32-row tiles), N=64 (NT=4), K=512, k' = (ky*4+kx)*32 + c
// ---------------------------------------------------------------------------
__global__ __launch_bounds__(32)
void conv2_kernel(const _Float16* __restrict__ in_all, const _Float16* __restrict__ wp,
                  const float* __restrict__ bias, _Float16* __restrict__ out) {
  const _Float16* in = in_all + (size_t)blockIdx.z * (512u * 400u * 32u);
  _Float16* o = out + (size_t)blockIdx.z * (512u * 81u * 64u);
  const int m0 = blockIdx.x * 32;
  wmma_gemm_tile<2, 4>(512,
      [&](int mrow, int k) -> const _Float16* {
        const int m = m0 + mrow;
        const int b = m / 81, rem = m % 81;
        const int oy = rem / 9, ox = rem % 9;
        const int pos = k >> 5, c = k & 31;
        const int ky = pos >> 2, kx = pos & 3;
        return in + (((size_t)b * 400 + (oy * 2 + ky) * 20 + (ox * 2 + kx)) * 32 + c);
      },
      [&](int n, int k) -> const _Float16* { return wp + (size_t)n * 512 + k; },
      [&](int mi, int n, float av) {
        const int m = m0 + mi;
        float val = av + bias[n];
        val = val > 0.f ? val : 0.f;
        o[(size_t)m * 64 + n] = (_Float16)val;
      });
}

// ---------------------------------------------------------------------------
// conv3: NHWC f16 [512,9,9,64], 3x3 s1 -> NHWC f16 [512,49,64]
// M=25088 (32-row tiles), N=64 (NT=4), K=576, k' = (ky*3+kx)*64 + c
// ---------------------------------------------------------------------------
__global__ __launch_bounds__(32)
void conv3_kernel(const _Float16* __restrict__ in_all, const _Float16* __restrict__ wp,
                  const float* __restrict__ bias, _Float16* __restrict__ out) {
  const _Float16* in = in_all + (size_t)blockIdx.z * (512u * 81u * 64u);
  _Float16* o = out + (size_t)blockIdx.z * (512u * 49u * 64u);
  const int m0 = blockIdx.x * 32;
  wmma_gemm_tile<2, 4>(576,
      [&](int mrow, int k) -> const _Float16* {
        const int m = m0 + mrow;
        const int b = m / 49, rem = m % 49;
        const int oy = rem / 7, ox = rem % 7;
        const int pos = k >> 6, c = k & 63;
        const int ky = pos / 3, kx = pos - ky * 3;
        return in + (((size_t)b * 81 + (oy + ky) * 9 + (ox + kx)) * 64 + c);
      },
      [&](int n, int k) -> const _Float16* { return wp + (size_t)n * 576 + k; },
      [&](int mi, int n, float av) {
        const int m = m0 + mi;
        float val = av + bias[n];
        val = val > 0.f ? val : 0.f;
        o[(size_t)m * 64 + n] = (_Float16)val;
      });
}

// ---------------------------------------------------------------------------
// FC: NHWC-flat f16 [512,3136] @ packed fcw [256][3136] + b, ReLU -> feat f16
// M=512 (32-row tiles), N=256 (4 y-blocks of NT=4), K=3136
// ---------------------------------------------------------------------------
__global__ __launch_bounds__(32)
void fc_kernel(const _Float16* __restrict__ in_all, const _Float16* __restrict__ wp,
               const float* __restrict__ bias, _Float16* __restrict__ out) {
  const _Float16* in = in_all + (size_t)blockIdx.z * (512u * 3136u);
  _Float16* o = out + (size_t)blockIdx.z * (512u * 256u);
  const int m0 = blockIdx.x * 32;
  const int n0 = blockIdx.y * 64;
  wmma_gemm_tile<2, 4>(3136,
      [&](int mrow, int k) -> const _Float16* { return in + (size_t)(m0 + mrow) * 3136 + k; },
      [&](int n, int k) -> const _Float16* { return wp + (size_t)(n0 + n) * 3136 + k; },
      [&](int mi, int n, float av) {
        float val = av + bias[n0 + n];
        val = val > 0.f ? val : 0.f;
        o[(size_t)(m0 + mi) * 256 + n0 + n] = (_Float16)val;
      });
}

// ---------------------------------------------------------------------------
// mlp1: z=0/1 relu(feat@e1w+e1b)->h1[z] ; z=2 relu(feat_next@pw1+pb1)->hp
// M=512 (32-row tiles), N=256, K=256
// ---------------------------------------------------------------------------
__global__ __launch_bounds__(32)
void mlp1_kernel(const _Float16* __restrict__ feat, const _Float16* __restrict__ e1wp,
                 const float* __restrict__ e1b, const _Float16* __restrict__ pw1p,
                 const float* __restrict__ pb1, _Float16* __restrict__ h1,
                 _Float16* __restrict__ hp) {
  const int z = blockIdx.z;
  const _Float16* in = feat + (size_t)((z == 0) ? 0 : 1) * (512u * 256u);
  const _Float16* wp = (z == 2) ? pw1p : e1wp;
  const float* bb    = (z == 2) ? pb1 : e1b;
  _Float16* o        = (z == 2) ? hp : (h1 + (size_t)z * (512u * 256u));
  const int m0 = blockIdx.x * 32;
  const int n0 = blockIdx.y * 64;
  wmma_gemm_tile<2, 4>(256,
      [&](int mrow, int k) -> const _Float16* { return in + (size_t)(m0 + mrow) * 256 + k; },
      [&](int n, int k) -> const _Float16* { return wp + (size_t)(n0 + n) * 256 + k; },
      [&](int mi, int n, float av) {
        float val = av + bb[n0 + n];
        val = val > 0.f ? val : 0.f;
        o[(size_t)(m0 + mi) * 256 + n0 + n] = (_Float16)val;
      });
}

// ---------------------------------------------------------------------------
// e2: phi = h1 @ e2w + e2b (no relu, fp32 out). M=512 (32-row), N=64, K=256
// ---------------------------------------------------------------------------
__global__ __launch_bounds__(32)
void e2_kernel(const _Float16* __restrict__ h1, const _Float16* __restrict__ wp,
               const float* __restrict__ bias, float* __restrict__ phi) {
  const _Float16* in = h1 + (size_t)blockIdx.z * (512u * 256u);
  float* o = phi + (size_t)blockIdx.z * (512u * 64u);
  const int m0 = blockIdx.x * 32;
  wmma_gemm_tile<2, 4>(256,
      [&](int mrow, int k) -> const _Float16* { return in + (size_t)(m0 + mrow) * 256 + k; },
      [&](int n, int k) -> const _Float16* { return wp + (size_t)n * 256 + k; },
      [&](int mi, int n, float av) {
        o[(size_t)(m0 + mi) * 64 + n] = av + bias[n];
      });
}

// ---------------------------------------------------------------------------
// c_y[b] = hp[b,:].pw2 + pb2  (N=1 -> VALU dot)
// ---------------------------------------------------------------------------
__global__ void cy_kernel(const _Float16* __restrict__ hp, const float* __restrict__ pw2,
                          const float* __restrict__ pb2, float* __restrict__ cy) {
  const int b = blockIdx.x * blockDim.x + threadIdx.x;
  if (b < 512) {
    float s = pb2[0];
#pragma unroll 8
    for (int k = 0; k < 256; ++k) s += (float)hp[(size_t)b * 256 + k] * pw2[k];
    cy[b] = s;
  }
}

// ---------------------------------------------------------------------------
// Fused pairwise MRN distance + row logsumexp + diag capture (logits never
// materialized). logits[i,j] = cy[j] - (max_d relu(x-y) + sqrt(sumsq + eps))
// ---------------------------------------------------------------------------
__global__ void row_lse_kernel(const float* __restrict__ phix,
                               const float* __restrict__ phiy,
                               const float* __restrict__ cy,
                               float* __restrict__ lse, float* __restrict__ dg) {
  __shared__ float xs[64];
  __shared__ float red[256];
  const int i = blockIdx.x;
  const int t = threadIdx.x;
  if (t < 64) xs[t] = phix[(size_t)i * 64 + t];
  __syncthreads();
  float lg[2];
  float lmax = -3.4e38f;
#pragma unroll
  for (int r = 0; r < 2; ++r) {
    const int j = t + r * 256;
    const float* y = phiy + (size_t)j * 64;
    float mx = 0.f;   // relu then max over non-neg values -> start at 0
#pragma unroll 8
    for (int d = 0; d < 32; ++d) mx = fmaxf(mx, xs[d] - y[d]);
    float ss = 0.f;
#pragma unroll 8
    for (int d = 32; d < 64; ++d) {
      const float diff = xs[d] - y[d];
      ss += diff * diff;
    }
    const float lgt = cy[j] - (mx + sqrtf(ss + 1e-8f));
    lg[r] = lgt;
    lmax = fmaxf(lmax, lgt);
    if (j == i) dg[i] = lgt;
  }
  red[t] = lmax;
  __syncthreads();
  for (int s = 128; s > 0; s >>= 1) {
    if (t < s) red[t] = fmaxf(red[t], red[t + s]);
    __syncthreads();
  }
  const float rmax = red[0];
  __syncthreads();
  red[t] = __expf(lg[0] - rmax) + __expf(lg[1] - rmax);
  __syncthreads();
  for (int s = 128; s > 0; s >>= 1) {
    if (t < s) red[t] += red[t + s];
    __syncthreads();
  }
  if (t == 0) lse[i] = rmax + __logf(red[0]);
}

// ---------------------------------------------------------------------------
// loss = mean(lse - diag) + 0.1 * mean((lse + 1e-6)^2)
// (logsumexp(logits + 1e-6) == logsumexp(logits) + 1e-6, uniform shift)
// ---------------------------------------------------------------------------
__global__ void final_kernel(const float* __restrict__ lse, const float* __restrict__ dg,
                             float* __restrict__ out) {
  __shared__ float r1[256], r2[256];
  const int t = threadIdx.x;
  float a = 0.f, b = 0.f;
#pragma unroll
  for (int r = 0; r < 2; ++r) {
    const int i = t + r * 256;
    const float l = lse[i];
    a += l - dg[i];
    const float l2 = l + 1e-6f;
    b += l2 * l2;
  }
  r1[t] = a; r2[t] = b;
  __syncthreads();
  for (int s = 128; s > 0; s >>= 1) {
    if (t < s) { r1[t] += r1[t + s]; r2[t] += r2[t + s]; }
    __syncthreads();
  }
  if (t == 0) out[0] = r1[0] / 512.f + 0.1f * (r2[0] / 512.f);
}

// ---------------------------------------------------------------------------
extern "C" void kernel_launch(void* const* d_in, const int* in_sizes, int n_in,
                              void* d_out, int out_size, void* d_ws, size_t ws_size,
                              hipStream_t stream) {
  const float* curr = (const float*)d_in[0];
  const float* next = (const float*)d_in[1];
  const float* c1w  = (const float*)d_in[2];
  const float* c1b  = (const float*)d_in[3];
  const float* c2w  = (const float*)d_in[4];
  const float* c2b  = (const float*)d_in[5];
  const float* c3w  = (const float*)d_in[6];
  const float* c3b  = (const float*)d_in[7];
  const float* fcw  = (const float*)d_in[8];
  const float* fcb  = (const float*)d_in[9];
  const float* e1w  = (const float*)d_in[10];
  const float* e1b  = (const float*)d_in[11];
  const float* e2w  = (const float*)d_in[12];
  const float* e2b  = (const float*)d_in[13];
  const float* pw1  = (const float*)d_in[14];
  const float* pb1  = (const float*)d_in[15];
  const float* pw2  = (const float*)d_in[16];
  const float* pb2  = (const float*)d_in[17];
  float* out = (float*)d_out;

  // --- workspace layout (fp32 region first, then 32B-aligned f16 region) ---
  float* phi = (float*)d_ws;          // 2*512*64 = 65,536 f32
  float* cy  = phi + 65536u;          // 512
  float* lse = cy + 512u;             // 512
  float* dgv = lse + 512u;            // 512
  _Float16* w1p  = (_Float16*)(dgv + 512u);  // 32*256        =      8,192 f16
  _Float16* w2p  = w1p + 8192u;              // 64*512        =     32,768
  _Float16* w3p  = w2p + 32768u;             // 64*576        =     36,864
  _Float16* fcwp = w3p + 36864u;             // 256*3136      =    802,816
  _Float16* e1wp = fcwp + 802816u;           // 256*256       =     65,536
  _Float16* pw1p = e1wp + 65536u;            // 256*256       =     65,536
  _Float16* e2wp = pw1p + 65536u;            // 64*256        =     16,384
  _Float16* inp  = e2wp + 16384u;            // 2*512*7056*4  = 28,901,376
  _Float16* out1 = inp + 28901376u;          // 2*512*400*32  = 13,107,200
  _Float16* out2 = out1 + 13107200u;         // 2*512*81*64   =  5,308,416
  _Float16* out3 = out2 + 5308416u;          // 2*512*49*64   =  3,211,264
  _Float16* feat = out3 + 3211264u;          // 2*512*256     =    262,144
  _Float16* h1   = feat + 262144u;           // 2*512*256     =    262,144
  _Float16* hp   = h1 + 262144u;             //   512*256     =    131,072

  // --- input + weight packing (f16; weights [N][K], conv K in (pos,c) order)
  pack_in_kernel<<<dim3(112896), 256, 0, stream>>>(curr, next, inp);
  pack_convw_kernel<<<dim3(32), 256, 0, stream>>>(c1w, w1p, 32, 4, 64);
  pack_convw_kernel<<<dim3(128), 256, 0, stream>>>(c2w, w2p, 64, 32, 16);
  pack_convw_kernel<<<dim3(144), 256, 0, stream>>>(c3w, w3p, 64, 64, 9);
  pack_fcw_kernel<<<dim3(3136), 256, 0, stream>>>(fcw, fcwp);
  packT_kernel<<<dim3(256), 256, 0, stream>>>(e1w, e1wp, 256, 256);
  packT_kernel<<<dim3(256), 256, 0, stream>>>(pw1, pw1p, 256, 256);
  packT_kernel<<<dim3(64), 256, 0, stream>>>(e2w, e2wp, 256, 64);

  // --- conv stack, both branches via blockIdx.z ---
  conv1_kernel<<<dim3(6400, 1, 2), 32, 0, stream>>>(inp, w1p, c1b, out1);
  conv2_kernel<<<dim3(1296, 1, 2), 32, 0, stream>>>(out1, w2p, c2b, out2);
  conv3_kernel<<<dim3(784, 1, 2), 32, 0, stream>>>(out2, w3p, c3b, out3);
  fc_kernel<<<dim3(16, 4, 2), 32, 0, stream>>>(out3, fcwp, fcb, feat);

  // --- heads ---
  mlp1_kernel<<<dim3(16, 4, 3), 32, 0, stream>>>(feat, e1wp, e1b, pw1p, pb1, h1, hp);
  e2_kernel<<<dim3(16, 1, 2), 32, 0, stream>>>(h1, e2wp, e2b, phi);
  cy_kernel<<<dim3(2), 256, 0, stream>>>(hp, pw2, pb2, cy);

  // --- fused pairwise distance + logsumexp + scalar loss ---
  row_lse_kernel<<<dim3(512), 256, 0, stream>>>(phi, phi + 512u * 64u, cy, lse, dgv);
  final_kernel<<<dim3(1), 256, 0, stream>>>(lse, dgv, out);
}
